// timeVae_3126736191846
// MI455X (gfx1250) — matrix operand order repaired and checked by reading, not compiled, not run on hardware
//
#include <hip/hip_runtime.h>

// ---------------- problem constants ----------------
#define TB   4
#define TL   2048
#define TD   512
#define THH  8
#define THD  64
#define NT   (TB * TL)     // 8192 tokens
#define DQKV (3 * TD)      // 1536

// ---------------- WMMA helper types ----------------
typedef __attribute__((ext_vector_type(16))) __bf16 bf16x16;
typedef __attribute__((ext_vector_type(8)))  float  f32x8;

union FragU { uint4 q[2]; bf16x16 v; };
union U4S  { uint4 q; unsigned short s[8]; };

__device__ __forceinline__ unsigned short f32_to_bf16(float f) {
  union { float f; unsigned int u; } x; x.f = f;
  unsigned int r = x.u + 0x7FFFu + ((x.u >> 16) & 1u);   // round-to-nearest-even
  return (unsigned short)(r >> 16);
}
__device__ __forceinline__ float silu_f(float v) { return v / (1.0f + __expf(-v)); }

__device__ __forceinline__ f32x8 zero8() {
  f32x8 z;
  #pragma unroll
  for (int i = 0; i < 8; ++i) z[i] = 0.0f;
  return z;
}

__device__ __forceinline__ f32x8 wmma_bf16(const FragU& a, const FragU& b, f32x8 c) {
  return __builtin_amdgcn_wmma_f32_16x16x32_bf16(false, a.v, false, b.v,
                                                 (short)0, c, false, false);
}

enum { GEMM_A_BF16 = 1, GEMM_A_SILU = 2, GEMM_OUT_BF16 = 4, GEMM_OUT_SILU = 8 };

// ---------------- generic bf16 WMMA GEMM:  C[M,N] = A[M,K] * W[N,K]^T + bias ----
// A: f32 (converted while staging, optional SiLU) or bf16 (flag)
// W: bf16 row-major [N][K];  C: f32 or bf16 (flag), optional SiLU epilogue
__global__ __launch_bounds__(256)
void gemm_wmma(const void* __restrict__ Aptr, const unsigned short* __restrict__ W,
               const float* __restrict__ bias, void* __restrict__ Cptr,
               int M, int N, int K, int flags)
{
  constexpr int BM = 128, BN = 64, BK = 64, PAD = 8;
  __shared__ alignas(16) unsigned short As[BM][BK + PAD];
  __shared__ alignas(16) unsigned short Bs[BN][BK + PAD];

  const int tid   = threadIdx.x;
  const int lane  = tid & 31;
  const int wave  = tid >> 5;
  const int wm    = wave & 3;         // 4 wave-rows  * 32 rows = 128
  const int wn    = wave >> 2;        // 2 wave-cols  * 32 cols = 64
  const int lrow  = lane & 15;
  const int lhalf = lane >> 4;
  const int m0 = blockIdx.x * BM, n0 = blockIdx.y * BN;

  f32x8 acc[2][2];
  #pragma unroll
  for (int i = 0; i < 2; ++i)
    #pragma unroll
    for (int j = 0; j < 2; ++j) acc[i][j] = zero8();

  for (int k0 = 0; k0 < K; k0 += BK) {
    // ---- stage A tile (f32->bf16 convert, optional SiLU; or raw bf16 copy)
    if (flags & GEMM_A_BF16) {
      const unsigned short* A = (const unsigned short*)Aptr;
      for (int idx = tid; idx < BM * BK; idx += 256) {
        int r = idx / BK, c = idx % BK;
        int gr = m0 + r, gc = k0 + c;
        unsigned short v = 0;
        if (gr < M && gc < K) v = A[(size_t)gr * K + gc];
        As[r][c] = v;
      }
    } else {
      const float* A = (const float*)Aptr;
      for (int idx = tid; idx < BM * BK; idx += 256) {
        int r = idx / BK, c = idx % BK;
        int gr = m0 + r, gc = k0 + c;
        float v = 0.0f;
        if (gr < M && gc < K) v = A[(size_t)gr * K + gc];
        if (flags & GEMM_A_SILU) v = silu_f(v);
        As[r][c] = f32_to_bf16(v);
      }
    }
    // ---- stage W tile (bf16 copy)
    for (int idx = tid; idx < BN * BK; idx += 256) {
      int r = idx / BK, c = idx % BK;
      int gn = n0 + r, gc = k0 + c;
      unsigned short v = 0;
      if (gn < N && gc < K) v = W[(size_t)gn * K + gc];
      Bs[r][c] = v;
    }
    // ---- software prefetch of next A tile (global_prefetch_b8)
    if (k0 + BK < K) {
      const char* p = (flags & GEMM_A_BF16)
        ? (const char*)((const unsigned short*)Aptr + (size_t)(m0 + (tid & 127)) * K + k0 + BK)
        : (const char*)((const float*)Aptr + (size_t)(m0 + (tid & 127)) * K + k0 + BK);
      __builtin_prefetch(p, 0, 1);
    }
    __syncthreads();

    #pragma unroll
    for (int ks = 0; ks < BK; ks += 32) {
      FragU aF[2], bF[2];
      #pragma unroll
      for (int i = 0; i < 2; ++i) {
        const unsigned short* aRow = &As[wm * 32 + i * 16 + lrow][0];
        aF[i].q[0] = *(const uint4*)(aRow + ks + lhalf * 8);        // K = half*8 + 0..7
        aF[i].q[1] = *(const uint4*)(aRow + ks + 16 + lhalf * 8);   // K = 16 + half*8 + 0..7
      }
      #pragma unroll
      for (int j = 0; j < 2; ++j) {
        const unsigned short* bRow = &Bs[wn * 32 + j * 16 + lrow][0];
        bF[j].q[0] = *(const uint4*)(bRow + ks + lhalf * 8);        // K = half*8 + 0..7
        bF[j].q[1] = *(const uint4*)(bRow + ks + 16 + lhalf * 8);   // K = 16 + half*8 + 0..7
      }
      #pragma unroll
      for (int i = 0; i < 2; ++i)
        #pragma unroll
        for (int j = 0; j < 2; ++j)
          acc[i][j] = wmma_bf16(aF[i], bF[j], acc[i][j]);
    }
    __syncthreads();
  }

  // ---- epilogue: C/D layout lane=N, vgpr r -> M = r + half*8
  #pragma unroll
  for (int i = 0; i < 2; ++i)
    #pragma unroll
    for (int j = 0; j < 2; ++j)
      #pragma unroll
      for (int r = 0; r < 8; ++r) {
        int row = m0 + wm * 32 + i * 16 + r + lhalf * 8;
        int col = n0 + wn * 32 + j * 16 + lrow;
        if (row < M && col < N) {
          float v = acc[i][j][r] + (bias ? bias[col] : 0.0f);
          if (flags & GEMM_OUT_SILU) v = silu_f(v);
          if (flags & GEMM_OUT_BF16)
            ((unsigned short*)Cptr)[(size_t)row * N + col] = f32_to_bf16(v);
          else
            ((float*)Cptr)[(size_t)row * N + col] = v;
        }
      }
}

// ---------------- flash attention ----------------
// qkv: bf16 [NT][1536] rows = token, cols: [Q(512) | K(512) | V(512)], head h at h*64.
// out: f32 [NT][512] (heads concatenated).
// grid = (L/64, H, B), block = 128 (4 waves). Each wave owns 16 query rows.
__global__ __launch_bounds__(128)
void flash_attn(const unsigned short* __restrict__ qkv, float* __restrict__ out)
{
  constexpr int KP = 32 + 8;
  __shared__ alignas(16) unsigned short Vt[4][THD][KP];  // per-wave V^T [d][key]
  __shared__ alignas(16) unsigned short Pt[4][16][KP];   // per-wave P   [qrow][key]

  const int lane  = threadIdx.x & 31;
  const int wave  = threadIdx.x >> 5;
  const int lrow  = lane & 15;
  const int lhalf = lane >> 4;
  const int b = blockIdx.z, h = blockIdx.y;
  const int q0 = blockIdx.x * 64 + wave * 16;
  const size_t RS = DQKV;                                // row stride in halfwords

  // Q fragments (A operand, 16x32 each), loaded once, fragment-layout direct from global
  FragU qF[2];
  {
    size_t qb = ((size_t)(b * TL + q0 + lrow)) * RS + (size_t)h * THD;
    #pragma unroll
    for (int i = 0; i < 2; ++i) {
      qF[i].q[0] = *(const uint4*)(qkv + qb + i * 32 + lhalf * 8);
      qF[i].q[1] = *(const uint4*)(qkv + qb + i * 32 + 16 + lhalf * 8);
    }
  }

  f32x8 o[4]; float mrow[8], lsum[8];
  #pragma unroll
  for (int f = 0; f < 4; ++f) o[f] = zero8();
  #pragma unroll
  for (int r = 0; r < 8; ++r) { mrow[r] = -1e30f; lsum[r] = 0.0f; }

  for (int kt = 0; kt < TL; kt += 32) {
    // stage V^T for this wave: lane stages key = lane
    {
      size_t vb = ((size_t)(b * TL + kt + lane)) * RS + 2 * TD + (size_t)h * THD;
      #pragma unroll
      for (int d0 = 0; d0 < THD; d0 += 8) {
        U4S u; u.q = *(const uint4*)(qkv + vb + d0);
        #pragma unroll
        for (int e = 0; e < 8; ++e) Vt[wave][d0 + e][lane] = u.s[e];
      }
    }
    // S = Q K^T (two key halves of 16, contraction over d=64 in two steps)
    f32x8 S[2];
    #pragma unroll
    for (int kf = 0; kf < 2; ++kf) {
      f32x8 a = zero8();
      size_t kb = ((size_t)(b * TL + kt + kf * 16 + lrow)) * RS + TD + (size_t)h * THD;
      #pragma unroll
      for (int ds = 0; ds < 2; ++ds) {
        FragU kB;
        kB.q[0] = *(const uint4*)(qkv + kb + ds * 32 + lhalf * 8);
        kB.q[1] = *(const uint4*)(qkv + kb + ds * 32 + 16 + lhalf * 8);
        a = wmma_bf16(qF[ds], kB, a);
      }
      S[kf] = a;
    }
    // online softmax (rows r + half*8, reduce over 16 lanes of the half)
    float psc[8];
    #pragma unroll
    for (int r = 0; r < 8; ++r) {
      float s0 = S[0][r] * 0.125f, s1 = S[1][r] * 0.125f;
      float v = fmaxf(s0, s1);
      #pragma unroll
      for (int m = 1; m <= 8; m <<= 1) v = fmaxf(v, __shfl_xor(v, m, 32));
      float mn = fmaxf(mrow[r], v);
      float sc = __expf(mrow[r] - mn);
      float p0 = __expf(s0 - mn), p1 = __expf(s1 - mn);
      S[0][r] = p0; S[1][r] = p1;
      float rs = p0 + p1;
      #pragma unroll
      for (int m = 1; m <= 8; m <<= 1) rs += __shfl_xor(rs, m, 32);
      lsum[r] = lsum[r] * sc + rs;
      mrow[r] = mn;
      psc[r] = sc;
    }
    #pragma unroll
    for (int f = 0; f < 4; ++f)
      #pragma unroll
      for (int r = 0; r < 8; ++r) o[f][r] *= psc[r];
    // transpose P through per-wave LDS (C-layout -> A-fragment layout)
    #pragma unroll
    for (int r = 0; r < 8; ++r) {
      int row = r + lhalf * 8;
      Pt[wave][row][lrow]      = f32_to_bf16(S[0][r]);
      Pt[wave][row][16 + lrow] = f32_to_bf16(S[1][r]);
    }
    FragU pA;
    pA.q[0] = *(const uint4*)&Pt[wave][lrow][lhalf * 8];
    pA.q[1] = *(const uint4*)&Pt[wave][lrow][16 + lhalf * 8];
    // O += P V   (4 d-blocks of 16)
    #pragma unroll
    for (int f = 0; f < 4; ++f) {
      FragU vB;
      vB.q[0] = *(const uint4*)&Vt[wave][f * 16 + lrow][lhalf * 8];
      vB.q[1] = *(const uint4*)&Vt[wave][f * 16 + lrow][16 + lhalf * 8];
      o[f] = wmma_bf16(pA, vB, o[f]);
    }
  }
  // normalize + store
  #pragma unroll
  for (int f = 0; f < 4; ++f)
    #pragma unroll
    for (int r = 0; r < 8; ++r) {
      int row = q0 + r + lhalf * 8;
      int col = h * THD + f * 16 + lrow;
      out[(size_t)(b * TL + row) * TD + col] = o[f][r] / lsum[r];
    }
}

// ---------------- residual + LayerNorm (one wave per row of D=512) -------------
__global__ __launch_bounds__(256)
void residual_layernorm(const float* __restrict__ A, const float* __restrict__ Bv,
                        const float* __restrict__ g, const float* __restrict__ be,
                        float* __restrict__ Y, int Mrows)
{
  const int lane = threadIdx.x & 31;
  const int wave = threadIdx.x >> 5;
  const int row = blockIdx.x * 8 + wave;
  if (row >= Mrows) return;
  const float* a = A + (size_t)row * TD;
  const float* b = Bv + (size_t)row * TD;
  float v[16], s = 0.0f;
  #pragma unroll
  for (int i = 0; i < 16; ++i) { v[i] = a[lane + i * 32] + b[lane + i * 32]; s += v[i]; }
  #pragma unroll
  for (int m = 1; m <= 16; m <<= 1) s += __shfl_xor(s, m, 32);
  float mean = s * (1.0f / TD), var = 0.0f;
  #pragma unroll
  for (int i = 0; i < 16; ++i) { float d = v[i] - mean; var += d * d; }
  #pragma unroll
  for (int m = 1; m <= 16; m <<= 1) var += __shfl_xor(var, m, 32);
  float inv = rsqrtf(var * (1.0f / TD) + 1e-5f);
  float* y = Y + (size_t)row * TD;
  #pragma unroll
  for (int i = 0; i < 16; ++i) {
    int c = lane + i * 32;
    y[c] = (v[i] - mean) * inv * g[c] + be[c];
  }
}

// ---------------- small elementwise / pack kernels ----------------
__global__ void pack_bf16(const float* __restrict__ w, unsigned short* __restrict__ o, int n) {
  int i = blockIdx.x * blockDim.x + threadIdx.x;
  if (i < n) o[i] = f32_to_bf16(w[i]);
}
// conv_w [out][in][3] -> packed [out][k*D + c]
__global__ void pack_conv(const float* __restrict__ w, unsigned short* __restrict__ o) {
  int i = blockIdx.x * blockDim.x + threadIdx.x;
  if (i >= TD * DQKV) return;
  int oc = i / DQKV, rem = i % DQKV, k = rem / TD, c = rem % TD;
  o[i] = f32_to_bf16(w[(size_t)oc * TD * 3 + c * 3 + k]);
}
// im2col with circular pad: Xim[b*L+l][k*D+c] = x[b][(l+k-1) mod L][c]
__global__ void im2col_circ(const float* __restrict__ x, unsigned short* __restrict__ Xim) {
  int i = blockIdx.x * blockDim.x + threadIdx.x;
  if (i >= NT * DQKV) return;
  int row = i / DQKV, col = i % DQKV, k = col / TD, c = col % TD;
  int b = row / TL, l = row % TL;
  int ls = l + k - 1; ls = (ls + TL) % TL;
  Xim[i] = f32_to_bf16(x[((size_t)(b * TL + ls)) * TD + c]);
}
// h = x + silu(emb + PE(l,d))
__global__ void emb_pos_silu(const float* __restrict__ x, const float* __restrict__ emb,
                             float* __restrict__ h) {
  int i = blockIdx.x * blockDim.x + threadIdx.x;
  if (i >= NT * TD) return;
  int d = i % TD, l = (i / TD) % TL;
  int de = d & ~1;
  float div = __expf(-(float)de * (9.210340371976184f / (float)TD));
  float pe = (d & 1) ? __cosf((float)l * div) : __sinf((float)l * div);
  float e = emb[i] + pe;
  h[i] = x[i] + silu_f(e);
}
__global__ void reparam(const float* __restrict__ mu, const float* __restrict__ lv,
                        const float* __restrict__ eps, float* __restrict__ z) {
  int i = blockIdx.x * blockDim.x + threadIdx.x;
  if (i >= NT * TD) return;
  z[i] = mu[i] + eps[i] * __expf(0.5f * lv[i]);
}

// ---------------- host side ----------------
struct EncW {
  const unsigned short *qkv, *out, *fc1, *fc2;
  const float *qkv_b, *out_b, *fc1_b, *fc2_b, *ln1_g, *ln1_b, *ln2_g, *ln2_b;
};

static void run_encoder(const float* X, float* Y,
                        unsigned short* qkvb, float* attn, float* proj, float* x1,
                        unsigned short* tmid, float* ffn, const EncW& w, hipStream_t s)
{
  dim3 gq(NT / 128, DQKV / 64), gd(NT / 128, TD / 64);
  gemm_wmma<<<gq, 256, 0, s>>>(X, w.qkv, w.qkv_b, qkvb, NT, DQKV, TD, GEMM_OUT_BF16);
  flash_attn<<<dim3(TL / 64, THH, TB), 128, 0, s>>>(qkvb, attn);
  gemm_wmma<<<gd, 256, 0, s>>>(attn, w.out, w.out_b, proj, NT, TD, TD, 0);
  residual_layernorm<<<NT / 8, 256, 0, s>>>(X, proj, w.ln1_g, w.ln1_b, x1, NT);
  gemm_wmma<<<gd, 256, 0, s>>>(x1, w.fc1, w.fc1_b, tmid, NT, TD, TD, GEMM_OUT_BF16 | GEMM_OUT_SILU);
  gemm_wmma<<<gd, 256, 0, s>>>(tmid, w.fc2, w.fc2_b, ffn, NT, TD, TD, GEMM_A_BF16);
  residual_layernorm<<<NT / 8, 256, 0, s>>>(x1, ffn, w.ln2_g, w.ln2_b, Y, NT);
}

extern "C" void kernel_launch(void* const* d_in, const int* in_sizes, int n_in,
                              void* d_out, int out_size, void* d_ws, size_t ws_size,
                              hipStream_t stream)
{
  (void)in_sizes; (void)n_in; (void)out_size; (void)ws_size;
  const float* x      = (const float*)d_in[0];
  const float* eps    = (const float*)d_in[1];
  const float* conv_w = (const float*)d_in[2];
  const float* conv_b = (const float*)d_in[3];
  const float* qkv_w  = (const float*)d_in[4];
  const float* qkv_b  = (const float*)d_in[5];
  const float* out_w  = (const float*)d_in[6];
  const float* out_b  = (const float*)d_in[7];
  const float* fc1_w  = (const float*)d_in[8];
  const float* fc1_b  = (const float*)d_in[9];
  const float* fc2_w  = (const float*)d_in[10];
  const float* fc2_b  = (const float*)d_in[11];
  const float* ln1_g  = (const float*)d_in[12];
  const float* ln1_b  = (const float*)d_in[13];
  const float* ln2_g  = (const float*)d_in[14];
  const float* ln2_b  = (const float*)d_in[15];
  const float* mean_w = (const float*)d_in[16];
  const float* mean_b = (const float*)d_in[17];
  const float* lv_w   = (const float*)d_in[18];
  const float* lv_b   = (const float*)d_in[19];
  const float* outp_w = (const float*)d_in[20];
  const float* outp_b = (const float*)d_in[21];

  float* outp = (float*)d_out;                 // [NT*TD]
  float* mu   = outp + (size_t)NT * TD;        // [NT*TD]
  float* lvar = outp + (size_t)2 * NT * TD;    // [NT*TD]

  char* base = (char*)d_ws; size_t off = 0;
  auto alloc = [&](size_t bytes) -> void* {
    void* p = base + off; off += (bytes + 255) & ~(size_t)255; return p;
  };
  unsigned short* wConv = (unsigned short*)alloc((size_t)TD * DQKV * 2);
  unsigned short* wQkv  = (unsigned short*)alloc((size_t)DQKV * TD * 2);
  unsigned short* wOut  = (unsigned short*)alloc((size_t)TD * TD * 2);
  unsigned short* wFc1  = (unsigned short*)alloc((size_t)TD * TD * 2);
  unsigned short* wFc2  = (unsigned short*)alloc((size_t)TD * TD * 2);
  unsigned short* wMean = (unsigned short*)alloc((size_t)TD * TD * 2);
  unsigned short* wLv   = (unsigned short*)alloc((size_t)TD * TD * 2);
  unsigned short* wOutp = (unsigned short*)alloc((size_t)TD * TD * 2);
  unsigned short* Xim   = (unsigned short*)alloc((size_t)NT * DQKV * 2);
  float* emb  = (float*)alloc((size_t)NT * TD * 4);
  float* hbuf = (float*)alloc((size_t)NT * TD * 4);
  unsigned short* qkvb = (unsigned short*)alloc((size_t)NT * DQKV * 2);
  float* attn = (float*)alloc((size_t)NT * TD * 4);
  float* proj = (float*)alloc((size_t)NT * TD * 4);
  float* x1   = (float*)alloc((size_t)NT * TD * 4);
  unsigned short* tmid = (unsigned short*)alloc((size_t)NT * TD * 2);
  float* ffn  = (float*)alloc((size_t)NT * TD * 4);
  float* enc1 = (float*)alloc((size_t)NT * TD * 4);
  float* zbuf = (float*)alloc((size_t)NT * TD * 4);
  float* enc2 = (float*)alloc((size_t)NT * TD * 4);

  auto cdiv = [](int a, int b) { return (a + b - 1) / b; };

  // weight packing (f32 -> bf16)
  pack_conv<<<cdiv(TD * DQKV, 256), 256, 0, stream>>>(conv_w, wConv);
  pack_bf16<<<cdiv(DQKV * TD, 256), 256, 0, stream>>>(qkv_w, wQkv, DQKV * TD);
  pack_bf16<<<cdiv(TD * TD, 256), 256, 0, stream>>>(out_w,  wOut,  TD * TD);
  pack_bf16<<<cdiv(TD * TD, 256), 256, 0, stream>>>(fc1_w,  wFc1,  TD * TD);
  pack_bf16<<<cdiv(TD * TD, 256), 256, 0, stream>>>(fc2_w,  wFc2,  TD * TD);
  pack_bf16<<<cdiv(TD * TD, 256), 256, 0, stream>>>(mean_w, wMean, TD * TD);
  pack_bf16<<<cdiv(TD * TD, 256), 256, 0, stream>>>(lv_w,   wLv,   TD * TD);
  pack_bf16<<<cdiv(TD * TD, 256), 256, 0, stream>>>(outp_w, wOutp, TD * TD);

  // token conv (im2col + WMMA GEMM) + positional + silu + residual
  im2col_circ<<<cdiv(NT * DQKV, 256), 256, 0, stream>>>(x, Xim);
  gemm_wmma<<<dim3(NT / 128, TD / 64), 256, 0, stream>>>(Xim, wConv, conv_b, emb,
                                                         NT, TD, DQKV, GEMM_A_BF16);
  emb_pos_silu<<<cdiv(NT * TD, 256), 256, 0, stream>>>(x, emb, hbuf);

  EncW ew{wQkv, wOut, wFc1, wFc2, qkv_b, out_b, fc1_b, fc2_b,
          ln1_g, ln1_b, ln2_g, ln2_b};

  // encoder pass 1
  run_encoder(hbuf, enc1, qkvb, attn, proj, x1, tmid, ffn, ew, stream);

  // mean / logvar heads straight into d_out
  dim3 gd(NT / 128, TD / 64);
  gemm_wmma<<<gd, 256, 0, stream>>>(enc1, wMean, mean_b, mu,   NT, TD, TD, 0);
  gemm_wmma<<<gd, 256, 0, stream>>>(enc1, wLv,   lv_b,   lvar, NT, TD, TD, 0);

  // reparameterize
  reparam<<<cdiv(NT * TD, 256), 256, 0, stream>>>(mu, lvar, eps, zbuf);

  // encoder pass 2 (shared weights)
  run_encoder(zbuf, enc2, qkvb, attn, proj, x1, tmid, ffn, ew, stream);

  // output head: out = silu(z2) @ outp_w^T + outp_b (SiLU fused into A staging)
  gemm_wmma<<<gd, 256, 0, stream>>>(enc2, wOutp, outp_b, outp, NT, TD, TD, GEMM_A_SILU);
}
